// VAE_SupervisedMixtureOfExperts_55362128445672
// MI455X (gfx1250) — matrix operand order repaired and checked by reading, not compiled, not run on hardware
//
#include <hip/hip_runtime.h>
#include <hip/hip_bf16.h>

typedef __attribute__((ext_vector_type(16))) _Float16 v16h;
typedef __attribute__((ext_vector_type(8)))  _Float16 v8h;
typedef __attribute__((ext_vector_type(8)))  float    v8f;

// ---------------------------------------------------------------------------
// Weight packing: produce per-lane WMMA B fragments, contiguous per lane.
// Packed layout: [e][chunk][colTile][lane(32)][16 halves]
// GEMM K index k = tap*Cin + ci  (tap-major so each 32-chunk has one tap)
// lane = (koff>>4)*16 + (n&15); idx = koff&15  (koff = k & 31)
// ---------------------------------------------------------------------------

// src: OIHW f32 [E][O][Cin][9]
__global__ void pack_conv_w_kernel(const float* __restrict__ src, _Float16* __restrict__ dst,
                                   int O, int Cin, int total) {
    int t = blockIdx.x * blockDim.x + threadIdx.x;
    if (t >= total) return;
    const int per = O * Cin * 9;
    int e = t / per;
    int r = t - e * per;
    int o = r / (Cin * 9);
    int r2 = r - o * (Cin * 9);
    int ci = r2 / 9;
    int tap = r2 - ci * 9;
    int k = tap * Cin + ci;
    int chunk = k >> 5;
    int koff = k & 31;
    int lane = ((koff >> 4) << 4) | (o & 15);
    int idx = koff & 15;
    int colTile = o >> 4;
    const int nCh = (Cin * 9) >> 5;
    const int nCT = O >> 4;
    size_t d = ((((size_t)e * nCh + chunk) * nCT + colTile) * 32 + lane) * 16 + idx;
    dst[d] = (_Float16)src[t];
}

// src: [Ksrc][N] f32. permute!=0 remaps GEMM row k -> src row (c*49 + p) with
// c = k & 127, p = k >> 7  (h3 is stored channel-last: kk = p*128 + c)
__global__ void pack_gemm_w_kernel(const float* __restrict__ src, _Float16* __restrict__ dst,
                                   int K, int N, int permute, int total) {
    int t = blockIdx.x * blockDim.x + threadIdx.x;
    if (t >= total) return;
    int k = t / N;
    int n = t - k * N;
    int srcRow = permute ? ((k & 127) * 49 + (k >> 7)) : k;
    int chunk = k >> 5;
    int koff = k & 31;
    int lane = ((koff >> 4) << 4) | (n & 15);
    int idx = koff & 15;
    int colTile = n >> 4;
    const int nCT = N >> 4;
    size_t d = (((size_t)chunk * nCT + colTile) * 32 + lane) * 16 + idx;
    dst[d] = (_Float16)src[(size_t)srcRow * N + n];
}

// ---------------------------------------------------------------------------
// Encoder conv1: 1->32, 28x28, pad 1, relu (K=9, VALU). NHWC f16 output.
// ---------------------------------------------------------------------------
__global__ void enc_conv1_kernel(const float* __restrict__ x, const float* __restrict__ w,
                                 const float* __restrict__ b, _Float16* __restrict__ out, int total) {
    int t = blockIdx.x * blockDim.x + threadIdx.x;
    if (t >= total) return;
    int c = t & 31;
    int p = (t >> 5) % 784;
    int bb = t / (784 * 32);
    int oy = p / 28, ox = p - oy * 28;
    float s = b[c];
    const float* xb = x + (size_t)bb * 784;
    const float* wc = w + c * 9;
    #pragma unroll
    for (int ky = 0; ky < 3; ++ky) {
        int iy = oy + ky - 1;
        if (iy < 0 || iy >= 28) continue;
        #pragma unroll
        for (int kx = 0; kx < 3; ++kx) {
            int ix = ox + kx - 1;
            if (ix < 0 || ix >= 28) continue;
            s += xb[iy * 28 + ix] * wc[ky * 3 + kx];
        }
    }
    out[((size_t)bb * 784 + p) * 32 + c] = (_Float16)fmaxf(s, 0.f);
}

// z = mu + eps * exp(0.5 * log_var)
__global__ void reparam_kernel(const float* __restrict__ mu, const float* __restrict__ lv,
                               const float* __restrict__ eps, float* __restrict__ z, int total) {
    int t = blockIdx.x * blockDim.x + threadIdx.x;
    if (t >= total) return;
    z[t] = mu[t] + eps[t] * __expf(0.5f * lv[t]);
}

// Gating MLP 64->64->32->8, softmax, logits, argmax. One wave per sample.
__global__ void gating_kernel(const float* __restrict__ z,
                              const float* __restrict__ gw1, const float* __restrict__ gb1,
                              const float* __restrict__ gw2, const float* __restrict__ gb2,
                              const float* __restrict__ gw3, const float* __restrict__ gb3,
                              float* __restrict__ out_logits, float* __restrict__ out_probs,
                              int* __restrict__ eidx) {
    __shared__ float sh[8][96];
    int wave = threadIdx.x >> 5;
    int lane = threadIdx.x & 31;
    int b = blockIdx.x * 8 + wave;
    const float* zb = z + (size_t)b * 64;
    #pragma unroll
    for (int u0 = 0; u0 < 2; ++u0) {
        int u = lane + 32 * u0;
        float s = gb1[u];
        for (int k = 0; k < 64; ++k) s += zb[k] * gw1[k * 64 + u];
        sh[wave][u] = fmaxf(s, 0.f);
    }
    __syncthreads();
    {
        float s = gb2[lane];
        for (int k = 0; k < 64; ++k) s += sh[wave][k] * gw2[k * 32 + lane];
        sh[wave][64 + lane] = fmaxf(s, 0.f);
    }
    __syncthreads();
    if (lane == 0) {
        float g[8];
        float mx = -1e30f;
        #pragma unroll
        for (int j = 0; j < 8; ++j) {
            float s = gb3[j];
            for (int k = 0; k < 32; ++k) s += sh[wave][64 + k] * gw3[k * 8 + j];
            g[j] = s;
            mx = fmaxf(mx, s);
        }
        float den = 0.f;
        #pragma unroll
        for (int j = 0; j < 8; ++j) { g[j] = __expf(g[j] - mx); den += g[j]; }
        float inv = 1.f / den;
        int best = 0;
        float bp = -1.f;
        #pragma unroll
        for (int j = 0; j < 8; ++j) {
            float p = g[j] * inv;
            out_probs[b * 8 + j]  = p;
            out_logits[b * 8 + j] = __logf(p + 1e-8f);
            if (p > bp) { bp = p; best = j; }
        }
        eidx[b] = best;
    }
}

// Decoder fc (selected expert): [64] -> [1568]; output NHWC [p(49)][c(32)]
__global__ void dec_fc_kernel(const float* __restrict__ z, const float* __restrict__ fw,
                              const float* __restrict__ fb, const int* __restrict__ eidx,
                              _Float16* __restrict__ d0, int total) {
    int t = blockIdx.x * blockDim.x + threadIdx.x;
    if (t >= total) return;
    int j = t % 1568;
    int b = t / 1568;
    int e = eidx[b];
    float s = fb[e * 1568 + j];
    const float* wf = fw + ((size_t)e * 64) * 1568 + j;
    const float* zb = z + (size_t)b * 64;
    #pragma unroll 8
    for (int k = 0; k < 64; ++k) s += zb[k] * wf[(size_t)k * 1568];
    int c = j / 49;
    int p = j - c * 49;
    d0[(size_t)b * 1568 + p * 32 + c] = (_Float16)s;
}

// Final decoder conv 32->1 + sigmoid (selected expert), NHWC input, f32 out
__global__ void dec_conv3_kernel(const _Float16* __restrict__ d2, const float* __restrict__ w3,
                                 const float* __restrict__ b3, const int* __restrict__ eidx,
                                 float* __restrict__ out, int total) {
    int t = blockIdx.x * blockDim.x + threadIdx.x;
    if (t >= total) return;
    int p = t % 784;
    int b = t / 784;
    int oy = p / 28, ox = p - oy * 28;
    int e = eidx[b];
    float s = b3[e];
    const _Float16* in = d2 + (size_t)b * 32 * 784;
    const float* w = w3 + e * 32 * 9;
    #pragma unroll
    for (int ky = 0; ky < 3; ++ky) {
        int iy = oy + ky - 1;
        if (iy < 0 || iy >= 28) continue;
        #pragma unroll
        for (int kx = 0; kx < 3; ++kx) {
            int ix = ox + kx - 1;
            if (ix < 0 || ix >= 28) continue;
            const _Float16* ip = in + (iy * 28 + ix) * 32;
            const float* wp = w + ky * 3 + kx;
            for (int ci = 0; ci < 32; ++ci)
                s += (float)ip[ci] * wp[ci * 9];
        }
    }
    out[t] = 1.f / (1.f + __expf(-s));
}

// ---------------------------------------------------------------------------
// WMMA implicit-GEMM convolution. NHWC f16 activations, packed f16 weights.
// rows = output pixels (per sample), cols = Cout, K = 9*Cin (tap-major).
// One sample per block; blockDim.x = 32*(Cout/16); wave w owns cols [16w,16w+16).
// upShift folds nearest 2x upsample into the gather. act: 0 none, 1 relu, 2 elu
// ---------------------------------------------------------------------------
__global__ void wmma_conv_kernel(const _Float16* __restrict__ in, int inSampleStride,
                                 int Cin, int Hin, int Win, int upShift, int stride,
                                 int Hout, int Wout,
                                 const _Float16* __restrict__ wPack, int wExpStride,
                                 const float* __restrict__ bias, int bExpStride,
                                 int Cout, _Float16* __restrict__ out, int outSampleStride,
                                 const int* __restrict__ expertIdx, int act) {
    const int sample = blockIdx.y;
    const int tileM = blockIdx.x;
    const int wave = threadIdx.x >> 5;
    const int lane = threadIdx.x & 31;
    const int laneHi = lane >> 4;
    const int lane15 = lane & 15;
    const int Hc = Hin << upShift, Wc = Win << upShift;
    const int HWout = Hout * Wout;
    const int m = tileM * 16 + lane15;
    const int oy = m / Wout, ox = m - (m / Wout) * Wout;
    const bool mValid = m < HWout;
    const int e = expertIdx ? expertIdx[sample] : 0;
    const int nCT = Cout >> 4;
    const int cinTiles = Cin >> 5;
    const _Float16* __restrict__ inS = in + (size_t)sample * inSampleStride;
    // per-lane base into packed weights for this wave's column tile
    const _Float16* __restrict__ Wl =
        wPack + (size_t)e * wExpStride + ((size_t)wave * 32 + lane) * 16;
    const size_t chunkStride = (size_t)nCT * 32 * 16;   // halves per K-chunk
    const v8h zero8 = {};

    v8f acc = {};
    int chunk = 0;
    for (int tap = 0; tap < 9; ++tap) {
        const int ky = tap / 3;
        const int kx = tap - ky * 3;
        const int iy = oy * stride + ky - 1;
        const int ix = ox * stride + kx - 1;
        const bool valid = mValid && iy >= 0 && iy < Hc && ix >= 0 && ix < Wc;
        int pix = 0;
        if (valid) pix = (iy >> upShift) * Win + (ix >> upShift);
        const _Float16* aBase = inS + (size_t)pix * Cin + laneHi * 16;
        for (int cb = 0; cb < cinTiles; ++cb, ++chunk) {
            const v8h* ap = (const v8h*)(aBase + (cb << 5));
            v8h a0 = ap[0];
            v8h a1 = ap[1];
            if (!valid) { a0 = zero8; a1 = zero8; }
            const v8h* bp = (const v8h*)(Wl + (size_t)chunk * chunkStride);
            v8h b0 = bp[0];
            v8h b1 = bp[1];
            v16h a = __builtin_shufflevector(a0, a1, 0, 1, 2, 3, 4, 5, 6, 7,
                                             8, 9, 10, 11, 12, 13, 14, 15);
            v16h b = __builtin_shufflevector(b0, b1, 0, 1, 2, 3, 4, 5, 6, 7,
                                             8, 9, 10, 11, 12, 13, 14, 15);
            acc = __builtin_amdgcn_wmma_f32_16x16x32_f16(false, a, false, b, (short)0, acc,
                                                         false, false);
        }
    }

    const int col = wave * 16 + lane15;
    const float bv = bias[e * bExpStride + col];
    #pragma unroll
    for (int v = 0; v < 8; ++v) {
        const int mm = tileM * 16 + v + laneHi * 8;
        if (mm < HWout) {
            float val = acc[v] + bv;
            if (act == 1) val = fmaxf(val, 0.f);
            else if (act == 2) val = (val > 0.f) ? val : expm1f(val);
            out[(size_t)sample * outSampleStride + (size_t)mm * Cout + col] = (_Float16)val;
        }
    }
}

// ---------------------------------------------------------------------------
// WMMA dense GEMM: out[Brows x N] = A[Brows x K] (f16 row-major) * packedW + bias
// grid.x = Brows/16, blockDim.x = 32*(N/16). f32 output.
// ---------------------------------------------------------------------------
__global__ void wmma_gemm_kernel(const _Float16* __restrict__ A, const _Float16* __restrict__ wPack,
                                 const float* __restrict__ bias, float* __restrict__ out,
                                 int K, int N) {
    const int tileM = blockIdx.x;
    const int wave = threadIdx.x >> 5;
    const int lane = threadIdx.x & 31;
    const int laneHi = lane >> 4;
    const int lane15 = lane & 15;
    const int nCT = N >> 4;
    const _Float16* __restrict__ aRow = A + (size_t)(tileM * 16 + lane15) * K + laneHi * 16;
    const _Float16* __restrict__ Wl = wPack + ((size_t)wave * 32 + lane) * 16;
    const size_t chunkStride = (size_t)nCT * 32 * 16;

    v8f acc = {};
    const int nChunks = K >> 5;
    for (int c = 0; c < nChunks; ++c) {
        const v8h* ap = (const v8h*)(aRow + (c << 5));
        v8h a0 = ap[0];
        v8h a1 = ap[1];
        const v8h* bp = (const v8h*)(Wl + (size_t)c * chunkStride);
        v8h b0 = bp[0];
        v8h b1 = bp[1];
        v16h a = __builtin_shufflevector(a0, a1, 0, 1, 2, 3, 4, 5, 6, 7,
                                         8, 9, 10, 11, 12, 13, 14, 15);
        v16h b = __builtin_shufflevector(b0, b1, 0, 1, 2, 3, 4, 5, 6, 7,
                                         8, 9, 10, 11, 12, 13, 14, 15);
        acc = __builtin_amdgcn_wmma_f32_16x16x32_f16(false, a, false, b, (short)0, acc,
                                                     false, false);
    }
    const int col = wave * 16 + lane15;
    const float bv = bias[col];
    #pragma unroll
    for (int v = 0; v < 8; ++v) {
        const int mm = tileM * 16 + v + laneHi * 8;
        out[(size_t)mm * N + col] = acc[v] + bv;
    }
}

// ---------------------------------------------------------------------------
// Launch
// ---------------------------------------------------------------------------
extern "C" void kernel_launch(void* const* d_in, const int* in_sizes, int n_in,
                              void* d_out, int out_size, void* d_ws, size_t ws_size,
                              hipStream_t stream) {
    const float* x      = (const float*)d_in[0];
    const float* eps    = (const float*)d_in[1];
    const float* enc_w1 = (const float*)d_in[2];
    const float* enc_b1 = (const float*)d_in[3];
    const float* enc_w2 = (const float*)d_in[4];
    const float* enc_b2 = (const float*)d_in[5];
    const float* enc_w3 = (const float*)d_in[6];
    const float* enc_b3 = (const float*)d_in[7];
    const float* w_mu   = (const float*)d_in[8];
    const float* b_mu   = (const float*)d_in[9];
    const float* w_lv   = (const float*)d_in[10];
    const float* b_lv   = (const float*)d_in[11];
    const float* g_w1   = (const float*)d_in[12];
    const float* g_b1   = (const float*)d_in[13];
    const float* g_w2   = (const float*)d_in[14];
    const float* g_b2   = (const float*)d_in[15];
    const float* g_w3   = (const float*)d_in[16];
    const float* g_b3   = (const float*)d_in[17];
    const float* d_fc_w = (const float*)d_in[18];
    const float* d_fc_b = (const float*)d_in[19];
    const float* d_w1   = (const float*)d_in[20];
    const float* d_b1   = (const float*)d_in[21];
    const float* d_w2   = (const float*)d_in[22];
    const float* d_b2   = (const float*)d_in[23];
    const float* d_w3   = (const float*)d_in[24];
    const float* d_b3   = (const float*)d_in[25];

    const int B = in_sizes[0] / 784;   // 2048

    float* out_rec    = (float*)d_out;
    float* out_mu     = out_rec + (size_t)B * 784;
    float* out_lv     = out_mu + (size_t)B * 64;
    float* out_logits = out_lv + (size_t)B * 64;
    float* out_probs  = out_logits + (size_t)B * 8;

    // ---- workspace layout ----
    char* ws = (char*)d_ws;
    size_t off = 0;
    auto take = [&](size_t bytes) -> char* {
        char* p = ws + off;
        off += (bytes + 255) & ~(size_t)255;
        return p;
    };
    _Float16* h1   = (_Float16*)take((size_t)B * 32 * 784 * 2);   // enc conv1 out (later: d2)
    _Float16* h2   = (_Float16*)take((size_t)B * 64 * 196 * 2);   // enc conv2 out (later: d1)
    _Float16* h3   = (_Float16*)take((size_t)B * 6272 * 2);       // enc conv3 out (later: d0)
    float*    zf   = (float*)take((size_t)B * 64 * 4);
    int*      eidx = (int*)take((size_t)B * 4);
    _Float16* w2p  = (_Float16*)take(288 * 64 * 2);
    _Float16* w3p  = (_Float16*)take(576 * 128 * 2);
    _Float16* wmup = (_Float16*)take(6272 * 64 * 2);
    _Float16* wlvp = (_Float16*)take(6272 * 64 * 2);
    _Float16* dw1p = (_Float16*)take((size_t)8 * 288 * 64 * 2);
    _Float16* dw2p = (_Float16*)take((size_t)8 * 576 * 32 * 2);
    _Float16* d0 = h3;   // reuse: h3 dead after mu/lv GEMM
    _Float16* d1 = h2;   // reuse: h2 dead after enc conv3
    _Float16* d2 = h1;   // reuse: h1 dead after enc conv2

    const int TB = 256;
    auto blocks = [](long n, int tb) { return (unsigned)((n + tb - 1) / tb); };

    // ---- weight prep: pack into per-lane WMMA B-fragment order ----
    pack_conv_w_kernel<<<blocks(64 * 32 * 9, TB), TB, 0, stream>>>(enc_w2, w2p, 64, 32,
                                                                   64 * 32 * 9);
    pack_conv_w_kernel<<<blocks(128 * 64 * 9, TB), TB, 0, stream>>>(enc_w3, w3p, 128, 64,
                                                                    128 * 64 * 9);
    pack_conv_w_kernel<<<blocks(8L * 64 * 32 * 9, TB), TB, 0, stream>>>(d_w1, dw1p, 64, 32,
                                                                        8 * 64 * 32 * 9);
    pack_conv_w_kernel<<<blocks(8L * 32 * 64 * 9, TB), TB, 0, stream>>>(d_w2, dw2p, 32, 64,
                                                                        8 * 32 * 64 * 9);
    pack_gemm_w_kernel<<<blocks(6272 * 64, TB), TB, 0, stream>>>(w_mu, wmup, 6272, 64, 1,
                                                                 6272 * 64);
    pack_gemm_w_kernel<<<blocks(6272 * 64, TB), TB, 0, stream>>>(w_lv, wlvp, 6272, 64, 1,
                                                                 6272 * 64);

    // ---- encoder ----
    enc_conv1_kernel<<<blocks((long)B * 32 * 784, TB), TB, 0, stream>>>(x, enc_w1, enc_b1, h1,
                                                                        B * 32 * 784);
    // conv2: 32->64, 28->14 (stride 2), relu. 196 rows -> 13 M-tiles, 4 waves
    wmma_conv_kernel<<<dim3(13, B), 128, 0, stream>>>(h1, 32 * 784, 32, 28, 28, 0, 2, 14, 14,
                                                      w2p, 0, enc_b2, 0, 64, h2, 64 * 196,
                                                      nullptr, 1);
    // conv3: 64->128, 14->7 (stride 2), relu. 49 rows -> 4 M-tiles, 8 waves
    wmma_conv_kernel<<<dim3(4, B), 256, 0, stream>>>(h2, 64 * 196, 64, 14, 14, 0, 2, 7, 7,
                                                     w3p, 0, enc_b3, 0, 128, h3, 128 * 49,
                                                     nullptr, 1);
    // mu / log_var projections: [B,6272] x [6272,64]
    wmma_gemm_kernel<<<B / 16, 128, 0, stream>>>(h3, wmup, b_mu, out_mu, 6272, 64);
    wmma_gemm_kernel<<<B / 16, 128, 0, stream>>>(h3, wlvp, b_lv, out_lv, 6272, 64);

    // ---- reparameterize + gating ----
    reparam_kernel<<<blocks((long)B * 64, TB), TB, 0, stream>>>(out_mu, out_lv, eps, zf, B * 64);
    gating_kernel<<<B / 8, 256, 0, stream>>>(zf, g_w1, g_b1, g_w2, g_b2, g_w3, g_b3,
                                             out_logits, out_probs, eidx);

    // ---- decoder (selected expert only; 8x FLOP reduction vs reference) ----
    dec_fc_kernel<<<blocks((long)B * 1568, TB), TB, 0, stream>>>(zf, d_fc_w, d_fc_b, eidx, d0,
                                                                 B * 1568);
    // dec conv1: up2(7->14) fused, 32->64, elu, per-expert weights
    wmma_conv_kernel<<<dim3(13, B), 128, 0, stream>>>(d0, 1568, 32, 7, 7, 1, 1, 14, 14,
                                                      dw1p, 288 * 64, d_b1, 64, 64, d1, 64 * 196,
                                                      eidx, 2);
    // dec conv2: up2(14->28) fused, 64->32, elu. 784 rows -> 49 M-tiles, 2 waves
    wmma_conv_kernel<<<dim3(49, B), 64, 0, stream>>>(d1, 64 * 196, 64, 14, 14, 1, 1, 28, 28,
                                                     dw2p, 576 * 32, d_b2, 32, 32, d2, 32 * 784,
                                                     eidx, 2);
    // dec conv3: 32->1 + sigmoid, straight to d_out
    dec_conv3_kernel<<<blocks((long)B * 784, TB), TB, 0, stream>>>(d2, d_w3, d_b3, eidx, out_rec,
                                                                   B * 784);
}
